// HGT_85452669322004
// MI455X (gfx1250) — matrix operand order repaired and checked by reading, not compiled, not run on hardware
//
#include <hip/hip_runtime.h>
#include <math.h>

typedef __attribute__((ext_vector_type(16))) _Float16 v16h;
typedef __attribute__((ext_vector_type(8)))  float    v8f;

#define HID 128

// ---------------- elementwise utility kernels ----------------
__global__ void k_fill(float* __restrict__ p, long n, float v) {
  long i = (long)blockIdx.x * blockDim.x + threadIdx.x;
  if (i < n) p[i] = v;
}

__global__ void k_gelu(float* __restrict__ p, long n) {
  long i = (long)blockIdx.x * blockDim.x + threadIdx.x;
  if (i < n) {
    float x = p[i];
    p[i] = 0.5f * x * (1.f + erff(x * 0.70710678118654752f));
  }
}

__global__ void k_blend(const float* __restrict__ o, const float* __restrict__ x,
                        const float* __restrict__ skip, float* __restrict__ y, long n) {
  long i = (long)blockIdx.x * blockDim.x + threadIdx.x;
  if (i < n) {
    float s = 1.f / (1.f + expf(-skip[0]));
    y[i] = s * o[i] + (1.f - s) * x[i];
  }
}

// Build block-diagonal 128x128 weight from rel[h, d, e] (H=2, D=64):
// Wblk[out = h*64+e][in = h*64+d] = rel[h][d][e]  (so Y = X @ Wblk^T == einsum nhd,hde->nhe)
__global__ void k_build_wblk(const float* __restrict__ rel, float* __restrict__ w) {
  int i = blockIdx.x * blockDim.x + threadIdx.x;  // 0..16383
  int o = i >> 7, in = i & 127;
  int ho = o >> 6, hi = in >> 6;
  w[i] = (ho == hi) ? rel[ho * 4096 + (in & 63) * 64 + (o & 63)] : 0.f;
}

// ---------------- WMMA GEMM: Y[M,128] = act(X[M,KIN] @ W[128,KIN]^T + b) ----------------
// Register-resident B: wave owns 16 rows x 64 cols (4 N tiles), holds all B frags
// for its N-half in VGPRs (KIN/32 x 4 x v16h), grid-strides over 64-row M chunks.
// Inner loop: A global loads + cvt, then 4*KIN/32 back-to-back v_wmma (no LDS).
template <int KIN>
__global__ __launch_bounds__(256)
void k_gemm_xwt(const float* __restrict__ X, const float* __restrict__ W,
                const float* __restrict__ bias, float* __restrict__ Y,
                int M, int relu, int nchunks)
{
  constexpr int KS = KIN / 32;
  const int tid  = threadIdx.x;
  const int wave = tid >> 5, lane = tid & 31;
  const int mw = wave >> 1;             // 0..3 : M subtile within 64-row chunk
  const int nh = wave & 1;              // 0..1 : N half (64 cols)
  const int bcol  = lane & 15;          // B frag: column within tile
  const int koffB = (lane >> 4) * 16;   // B frag: K offset
  const int koffA = (lane >> 4) * 16;   // A frag: K offset

  // B fragments, register resident (f32 -> f16 convert from global; W is L2-resident)
  v16h B[KS][4];
#pragma unroll
  for (int ks = 0; ks < KS; ++ks) {
#pragma unroll
    for (int nt = 0; nt < 4; ++nt) {
      const float* wrow = W + (size_t)(nh * 64 + nt * 16 + bcol) * KIN + ks * 32 + koffB;
#pragma unroll
      for (int j = 0; j < 16; ++j) B[ks][nt][j] = (_Float16)wrow[j];
    }
  }

  const int crow = (lane >> 4) * 8;     // C/D layout: VGPR i -> M = i + 8*(lane/16)
  const int ccol = lane & 15;
  float bv[4];
#pragma unroll
  for (int nt = 0; nt < 4; ++nt)
    bv[nt] = bias ? bias[nh * 64 + nt * 16 + ccol] : 0.f;

  for (int chunk = blockIdx.x; chunk < nchunks; chunk += gridDim.x) {
    const int row0 = chunk * 64 + mw * 16;
    int arow = row0 + (lane & 15);
    if (arow >= M) arow = M - 1;        // clamp; stores guarded (EXEC all-1s for WMMA)
    const float* xrow = X + (size_t)arow * KIN;

    v16h A[KS];
#pragma unroll
    for (int ks = 0; ks < KS; ++ks) {
#pragma unroll
      for (int j = 0; j < 8; ++j) {
        A[ks][j]     = (_Float16)xrow[ks * 32 + koffA + j];
        A[ks][8 + j] = (_Float16)xrow[ks * 32 + koffA + 8 + j];
      }
    }

    v8f acc[4] = {};
#pragma unroll
    for (int ks = 0; ks < KS; ++ks)
#pragma unroll
      for (int nt = 0; nt < 4; ++nt)
        acc[nt] = __builtin_amdgcn_wmma_f32_16x16x32_f16(
            false, A[ks], false, B[ks][nt], (short)0, acc[nt], false, false);

#pragma unroll
    for (int nt = 0; nt < 4; ++nt) {
#pragma unroll
      for (int i = 0; i < 8; ++i) {
        int r = row0 + crow + i;
        if (r < M) {
          float v = acc[nt][i] + bv[nt];
          if (relu) v = v > 0.f ? v : 0.f;
          Y[(size_t)r * HID + nh * 64 + nt * 16 + ccol] = v;
        }
      }
    }
  }
}

// ---------------- edge-phase kernels (wave32 per edge) ----------------
__device__ inline void atomicMaxF(float* addr, float val) {
  unsigned* u = (unsigned*)addr;
  unsigned old = *u;
  while (__uint_as_float(old) < val) {
    unsigned assumed = old;
    old = atomicCAS(u, assumed, __float_as_uint(val));
    if (old == assumed) break;
  }
}

// alpha[e,h] = (q[dst,h,:] . kr[src,h,:]) * p_rel[h] / sqrt(64); track per-dst max.
// One b128 load per lane per row: lanes 0..15 -> head 0, lanes 16..31 -> head 1.
__global__ __launch_bounds__(256)
void k_edge_alpha(const float* __restrict__ q, const float* __restrict__ kr,
                  const int* __restrict__ src, const int* __restrict__ dst,
                  const float* __restrict__ p_rel,
                  float* __restrict__ alpha, float* __restrict__ amax, int E)
{
  int e    = (blockIdx.x * 256 + threadIdx.x) >> 5;
  int lane = threadIdx.x & 31;
  if (e >= E) return;
  int s = src[e], d = dst[e];
  float4 qv = ((const float4*)(q  + (size_t)d * HID))[lane];
  float4 kv = ((const float4*)(kr + (size_t)s * HID))[lane];
  float v = qv.x * kv.x + qv.y * kv.y + qv.z * kv.z + qv.w * kv.w;
#pragma unroll
  for (int off = 8; off > 0; off >>= 1) v += __shfl_xor(v, off, 32);  // reduce within 16-lane halves
  int h = lane >> 4;
  if ((lane & 15) == 0) {
    float a = v * p_rel[h] * 0.125f;
    alpha[(size_t)e * 2 + h] = a;
    atomicMaxF(&amax[(size_t)d * 2 + h], a);
  }
}

__global__ void k_edge_expsum(float* __restrict__ alpha, const float* __restrict__ amax,
                              float* __restrict__ asum, const int* __restrict__ dst, int E)
{
  long i = (long)blockIdx.x * blockDim.x + threadIdx.x;
  if (i >= (long)E * 2) return;
  int e = (int)(i >> 1), h = (int)(i & 1);
  int d = dst[e];
  float a = expf(alpha[i] - amax[(size_t)d * 2 + h]);
  alpha[i] = a;
  atomicAdd(&asum[(size_t)d * 2 + h], a);
}

__global__ __launch_bounds__(256)
void k_edge_scatter(const float* __restrict__ alpha, const float* __restrict__ asum,
                    const float* __restrict__ vr, float* __restrict__ out,
                    const int* __restrict__ src, const int* __restrict__ dst, int E)
{
  int e    = (blockIdx.x * 256 + threadIdx.x) >> 5;
  int lane = threadIdx.x & 31;
  if (e >= E) return;
  int s = src[e], d = dst[e];
  float4 vv = ((const float4*)(vr + (size_t)s * HID))[lane];
  int h = lane >> 4;
  float w = alpha[(size_t)e * 2 + h] / (asum[(size_t)d * 2 + h] + 1e-16f);
  float* op = out + (size_t)d * HID + lane * 4;
  atomicAdd(op + 0, vv.x * w);
  atomicAdd(op + 1, vv.y * w);
  atomicAdd(op + 2, vv.z * w);
  atomicAdd(op + 3, vv.w * w);
}

// ---------------- host orchestration ----------------
extern "C" void kernel_launch(void* const* d_in, const int* in_sizes, int n_in,
                              void* d_out, int out_size, void* d_ws, size_t ws_size,
                              hipStream_t stream)
{
  (void)in_sizes; (void)n_in; (void)out_size; (void)ws_size;
  constexpr int NA = 100000, NB = 100000, E = 800000, L = 2;

  const float* x_a = (const float*)d_in[0];
  const float* x_b = (const float*)d_in[1];
  const int*  e_ab = (const int*)d_in[2];   // [0..E): src(a), [E..2E): dst(b)
  const int*  e_ba = (const int*)d_in[3];   // [0..E): src(b), [E..2E): dst(a)
  auto F = [&](int i) { return (const float*)d_in[i]; };

  // ---- workspace carve ----
  char* wp = (char*)d_ws;
  auto carve = [&](size_t nf) {
    float* r = (float*)wp;
    wp += ((nf * sizeof(float) + 255) & ~(size_t)255);
    return r;
  };
  const size_t SA = (size_t)NA * HID, SB = (size_t)NB * HID;
  const size_t SMAX = SA > SB ? SA : SB;
  float* hA   = carve(SA);
  float* hB   = carve(SB);
  float* qA   = carve(SA);
  float* qB   = carve(SB);
  float* krA  = carve(SA);
  float* vrA  = carve(SA);
  float* krB  = carve(SB);
  float* vrB  = carve(SB);
  float* tmp  = carve(SMAX);
  float* outA = carve(SA);
  float* outB = carve(SB);
  float* alph = carve((size_t)E * 2);
  float* amax = carve((size_t)(NA > NB ? NA : NB) * 2);
  float* asum = carve((size_t)(NA > NB ? NA : NB) * 2);
  float* wblk = carve(128 * 128);

  auto gemm = [&](const float* X, const float* W, const float* b, float* Y,
                  int M, int Kin, int relu) {
    int nch = (M + 63) / 64;
    int grid = nch < 1536 ? nch : 1536;   // persistent grid-stride blocks
    if (Kin == 128)
      k_gemm_xwt<128><<<dim3(grid), dim3(256), 0, stream>>>(X, W, b, Y, M, relu, nch);
    else if (Kin == 64)
      k_gemm_xwt<64><<<dim3(grid), dim3(256), 0, stream>>>(X, W, b, Y, M, relu, nch);
    else
      k_gemm_xwt<32><<<dim3(grid), dim3(256), 0, stream>>>(X, W, b, Y, M, relu, nch);
  };
  auto fill = [&](float* p, long n, float v) {
    k_fill<<<dim3((unsigned)((n + 255) / 256)), dim3(256), 0, stream>>>(p, n, v);
  };
  const unsigned EB = (E + 7) / 8;  // wave-per-edge blocks

  // ---- input projections + relu ----
  gemm(x_a, F(4), F(5), hA, NA, 64, 1);
  gemm(x_b, F(6), F(7), hB, NB, 32, 1);

  for (int layer = 0; layer < L; ++layer) {
    const int L0 = 8 + layer * 24;
    const float *kaW = F(L0+0),  *kab = F(L0+1),  *kbW = F(L0+2),  *kbb = F(L0+3);
    const float *qaW = F(L0+4),  *qab = F(L0+5),  *qbW = F(L0+6),  *qbb = F(L0+7);
    const float *vaW = F(L0+8),  *vab = F(L0+9),  *vbW = F(L0+10), *vbb = F(L0+11);
    const float *aaW = F(L0+12), *aab = F(L0+13), *abW = F(L0+14), *abb = F(L0+15);
    const float *skA = F(L0+16), *skB = F(L0+17);
    const float *ab_a = F(L0+18), *ab_m = F(L0+19), *ab_p = F(L0+20);
    const float *ba_a = F(L0+21), *ba_m = F(L0+22), *ba_p = F(L0+23);

    // q projections
    gemm(hA, qaW, qab, qA, NA, HID, 0);
    gemm(hB, qbW, qbb, qB, NB, HID, 0);
    // k/v projections + relation transforms (block-diagonal WMMA GEMM)
    gemm(hA, kaW, kab, tmp, NA, HID, 0);
    k_build_wblk<<<64, 256, 0, stream>>>(ab_a, wblk);
    gemm(tmp, wblk, nullptr, krA, NA, HID, 0);
    gemm(hA, vaW, vab, tmp, NA, HID, 0);
    k_build_wblk<<<64, 256, 0, stream>>>(ab_m, wblk);
    gemm(tmp, wblk, nullptr, vrA, NA, HID, 0);
    gemm(hB, kbW, kbb, tmp, NB, HID, 0);
    k_build_wblk<<<64, 256, 0, stream>>>(ba_a, wblk);
    gemm(tmp, wblk, nullptr, krB, NB, HID, 0);
    gemm(hB, vbW, vbb, tmp, NB, HID, 0);
    k_build_wblk<<<64, 256, 0, stream>>>(ba_m, wblk);
    gemm(tmp, wblk, nullptr, vrB, NB, HID, 0);

    fill(outA, (long)SA, 0.f);
    fill(outB, (long)SB, 0.f);

    // edge type "ab": a -> b (dst = b, uses qB, krA, vrA)
    fill(amax, (long)NB * 2, -INFINITY);
    fill(asum, (long)NB * 2, 0.f);
    k_edge_alpha<<<EB, 256, 0, stream>>>(qB, krA, e_ab, e_ab + E, ab_p, alph, amax, E);
    k_edge_expsum<<<(unsigned)(((long)E * 2 + 255) / 256), 256, 0, stream>>>(alph, amax, asum, e_ab + E, E);
    k_edge_scatter<<<EB, 256, 0, stream>>>(alph, asum, vrA, outB, e_ab, e_ab + E, E);

    // edge type "ba": b -> a (dst = a, uses qA, krB, vrB)
    fill(amax, (long)NA * 2, -INFINITY);
    fill(asum, (long)NA * 2, 0.f);
    k_edge_alpha<<<EB, 256, 0, stream>>>(qA, krB, e_ba, e_ba + E, ba_p, alph, amax, E);
    k_edge_expsum<<<(unsigned)(((long)E * 2 + 255) / 256), 256, 0, stream>>>(alph, amax, asum, e_ba + E, E);
    k_edge_scatter<<<EB, 256, 0, stream>>>(alph, asum, vrB, outA, e_ba, e_ba + E, E);

    // output stage: gelu -> a-linear -> sigmoid-skip blend
    float* yA = (layer == L - 1) ? (float*)d_out : hA;
    float* yB = (layer == L - 1) ? ((float*)d_out + SA) : hB;

    k_gelu<<<(unsigned)((SA + 255) / 256), 256, 0, stream>>>(outA, (long)SA);
    gemm(outA, aaW, aab, tmp, NA, HID, 0);
    k_blend<<<(unsigned)((SA + 255) / 256), 256, 0, stream>>>(tmp, hA, skA, yA, (long)SA);

    k_gelu<<<(unsigned)((SB + 255) / 256), 256, 0, stream>>>(outB, (long)SB);
    gemm(outB, abW, abb, tmp, NB, HID, 0);
    k_blend<<<(unsigned)((SB + 255) / 256), 256, 0, stream>>>(tmp, hB, skB, yB, (long)SB);
  }
}
